// gru_26001732010555
// MI455X (gfx1250) — matrix operand order repaired and checked by reading, not compiled, not run on hardware
//
#include <hip/hip_runtime.h>
#include <hip/hip_bf16.h>
#include <math.h>

// ---------------------------------------------------------------------------
// Problem constants (from reference): S=50, B=128, D=512, H=512, V=30000
// ---------------------------------------------------------------------------
#define S_LEN 50
#define BATCH 128
#define DDIM  512
#define HDIM  512
#define H3    1536
#define VOCAB 30000
#define SB    (S_LEN * BATCH)      // 6400 rows
#define NSPLIT 25                  // vocab column splits for fused loss
#define TPS    75                  // 16-col tiles per split: 25*75*16 = 30000
#define RG     8                   // row-blocks (waves) per workgroup in loss
#define LOSS_THREADS (RG * 32)     // 256
#define TILE_BYTES (16 * HDIM * 2) // 16KB staged W_out tile

typedef __attribute__((ext_vector_type(16))) __bf16 v16bf;
typedef __attribute__((ext_vector_type(8)))  float  v8f;

// ---------------------------------------------------------------------------
// WMMA fragment loaders (bf16 16x16x32, wave32, contiguous k order used for both A and B)
// A (16x32, row-major, lda elems):
//   lane 0-15 : M=lane,    elems 0-15 -> K=k0+0..15
//   lane16-31 : M=lane-16, elems 0-15 -> K=k0+16..31
// B (32x16 = rows of weight[N][K], ldb elems):
//   lane 0-15 : N=lane,    elems 0-15 -> K=k0+0..15
//   lane16-31 : N=lane-16, elems 0-15 -> K=k0+16..31
// ---------------------------------------------------------------------------
__device__ __forceinline__ v16bf load_a16(const __bf16* A, int lda, int row0,
                                          int k0, int lane) {
  int m  = row0 + (lane & 15);
  int kb = (lane & 16) ? 16 : 0;
  const __bf16* p = A + (size_t)m * lda + k0 + kb;
  union { float4 q; __bf16 h[8]; } lo, hi;
  lo.q = *(const float4*)(p);
  hi.q = *(const float4*)(p + 8);
  v16bf a;
#pragma unroll
  for (int i = 0; i < 8; ++i) { a[i] = lo.h[i]; a[i + 8] = hi.h[i]; }
  return a;
}

__device__ __forceinline__ v16bf load_b16(const __bf16* Bt, int ldb, int col0,
                                          int k0, int lane) {
  int n  = col0 + (lane & 15);
  int kb = (lane & 16) ? 16 : 0;
  const __bf16* p = Bt + (size_t)n * ldb + k0 + kb;
  union { float4 q; __bf16 h[8]; } lo, hi;
  lo.q = *(const float4*)(p);
  hi.q = *(const float4*)(p + 8);
  v16bf b;
#pragma unroll
  for (int i = 0; i < 8; ++i) { b[i] = lo.h[i]; b[i + 8] = hi.h[i]; }
  return b;
}

// B fragment sourced from an LDS-staged tile Bs[16][512] (row-major).
__device__ __forceinline__ v16bf load_b16_lds(const __bf16* Bs, int k0, int lane) {
  int n  = lane & 15;
  int kb = (lane & 16) ? 16 : 0;
  const __bf16* p = Bs + (size_t)n * HDIM + k0 + kb;
  union { float4 q; __bf16 h[8]; } lo, hi;
  lo.q = *(const float4*)(p);
  hi.q = *(const float4*)(p + 8);
  v16bf b;
#pragma unroll
  for (int i = 0; i < 8; ++i) { b[i] = lo.h[i]; b[i + 8] = hi.h[i]; }
  return b;
}

// ---------------------------------------------------------------------------
// CDNA5 async copy: stage one contiguous 16KB tile (16 vocab rows x 512 bf16)
// global -> LDS. LOSS_THREADS x 16B x 4 iters. Tracked by ASYNCcnt.
// ---------------------------------------------------------------------------
__device__ __forceinline__ void async_stage_tile(const __bf16* gsrc,
                                                 unsigned lds_base, int tid) {
#pragma unroll
  for (int it = 0; it < TILE_BYTES / (LOSS_THREADS * 16); ++it) {
    unsigned off = (unsigned)(it * (LOSS_THREADS * 16) + tid * 16);
    unsigned long long ga = (unsigned long long)(const char*)gsrc + off;
    unsigned la = lds_base + off;
    asm volatile("global_load_async_to_lds_b128 %0, %1, off"
                 :: "v"(la), "v"(ga) : "memory");
  }
}

__device__ __forceinline__ void wait_async0() {
  asm volatile("s_wait_asynccnt 0x0" ::: "memory");
}

// ---------------------------------------------------------------------------
// Utility kernels: f32->bf16 convert, embedding gather, zero-fill
// ---------------------------------------------------------------------------
__global__ void cvt_bf16_kernel(const float* __restrict__ src,
                                __bf16* __restrict__ dst, long n) {
  long i = (long)blockIdx.x * blockDim.x + threadIdx.x;
  if (i < n) dst[i] = (__bf16)src[i];
}

__global__ void embed_kernel(const int* __restrict__ ids,
                             const float* __restrict__ emb,
                             __bf16* __restrict__ x) {
  long i = (long)blockIdx.x * blockDim.x + threadIdx.x;   // over SB*D
  if (i >= (long)SB * DDIM) return;
  int row = (int)(i / DDIM);
  int d   = (int)(i % DDIM);
  x[i] = (__bf16)emb[(size_t)ids[row] * DDIM + d];
}

__global__ void zero_h_kernel(float* __restrict__ h, __bf16* __restrict__ hb) {
  int i = blockIdx.x * blockDim.x + threadIdx.x;
  if (i < BATCH * HDIM) { h[i] = 0.0f; hb[i] = (__bf16)0.0f; }
}

// ---------------------------------------------------------------------------
// Generic bf16 WMMA GEMM: C[M,N] = A[M,K] @ Bt[N,K]^T + bias[N]
// One wave per 16x16 tile. (Used for the hoisted GI = X @ W_ih^T + b_ih.)
// ---------------------------------------------------------------------------
__global__ void gemm_bias_kernel(const __bf16* __restrict__ A,
                                 const __bf16* __restrict__ Bt,
                                 const float*  __restrict__ bias,
                                 float* __restrict__ C,
                                 int M, int N, int K, int NT /* = N/16 */) {
  int lane = threadIdx.x & 31;
  int gw   = blockIdx.x * (blockDim.x >> 5) + (threadIdx.x >> 5);
  int tm   = gw / NT;
  int tn   = gw % NT;
  if (tm * 16 >= M) return;                 // whole wave exits together

  v8f acc = {};
  for (int k0 = 0; k0 < K; k0 += 32) {
    v16bf a = load_a16(A, K, tm * 16, k0, lane);
    v16bf b = load_b16(Bt, K, tn * 16, k0, lane);
    acc = __builtin_amdgcn_wmma_f32_16x16x32_bf16(false, a, false, b,
                                                  (short)0, acc, false, false);
  }
  int n  = tn * 16 + (lane & 15);
  float bo = bias[n];
  int mh = (lane & 16) ? 8 : 0;
#pragma unroll
  for (int r = 0; r < 8; ++r) {
    int m = tm * 16 + mh + r;
    C[(size_t)m * N + n] = acc[r] + bo;
  }
}

// ---------------------------------------------------------------------------
// Fused GRU step: one block = 96 threads = 3 waves = the 3 gates of one
// 16(batch) x 16(hidden) tile. Wave g computes the GH tile for gate g via
// WMMA, tiles meet in LDS, gates applied in-block. h is double-buffered.
// Grid: (BATCH/16) * (HDIM/16) = 8*32 = 256 blocks.
// ---------------------------------------------------------------------------
__global__ void gru_step_kernel(const float*  __restrict__ GI,       // [SB,3H]
                                const __bf16* __restrict__ Whh_bf,   // [3H,H]
                                const float*  __restrict__ bhh,      // [3H]
                                const float*  __restrict__ h_prev,   // [B,H]
                                const __bf16* __restrict__ hbf_prev, // [B,H]
                                float*  __restrict__ h_out,          // [B,H]
                                __bf16* __restrict__ hbf_out,        // [B,H]
                                __bf16* __restrict__ hs_bf,          // [SB,H]
                                int t) {
  __shared__ float Gs[3][16][16];
  int tid  = threadIdx.x;
  int lane = tid & 31;
  int g    = tid >> 5;                       // gate index 0..2 (wave id)
  int tm   = blockIdx.x / (HDIM / 16);       // batch tile
  int tj   = blockIdx.x % (HDIM / 16);       // hidden tile
  int n0   = g * HDIM + tj * 16;             // column base in [3H]

  v8f acc = {};
  for (int k0 = 0; k0 < HDIM; k0 += 32) {
    v16bf a = load_a16(hbf_prev, HDIM, tm * 16, k0, lane);
    v16bf b = load_b16(Whh_bf, HDIM, n0, k0, lane);
    acc = __builtin_amdgcn_wmma_f32_16x16x32_bf16(false, a, false, b,
                                                  (short)0, acc, false, false);
  }
  int n  = lane & 15;
  int mh = (lane & 16) ? 8 : 0;
  float bo = bhh[n0 + n];
#pragma unroll
  for (int r = 0; r < 8; ++r) Gs[g][mh + r][n] = acc[r] + bo;
  __syncthreads();

  for (int i = tid; i < 256; i += 96) {
    int rr = i >> 4, cc = i & 15;
    int b  = tm * 16 + rr;
    int j  = tj * 16 + cc;
    const float* gi = GI + ((size_t)t * BATCH + b) * H3;
    float ir = gi[j], iz = gi[j + HDIM], in_ = gi[j + 2 * HDIM];
    float hr = Gs[0][rr][cc], hz = Gs[1][rr][cc], hn = Gs[2][rr][cc];
    float rg = 1.0f / (1.0f + __expf(-(ir + hr)));
    float zg = 1.0f / (1.0f + __expf(-(iz + hz)));
    float ng = tanhf(in_ + rg * hn);
    float hp = h_prev[(size_t)b * HDIM + j];
    float hv = (1.0f - zg) * ng + zg * hp;
    h_out[(size_t)b * HDIM + j] = hv;
    __bf16 hb = (__bf16)hv;
    hbf_out[(size_t)b * HDIM + j] = hb;
    hs_bf[((size_t)t * BATCH + b) * HDIM + j] = hb;
  }
}

// ---------------------------------------------------------------------------
// Fused vocab GEMM + online logsumexp + target gather (logits never stored).
// Block = 8 waves = 8 row-blocks sharing one vocab split. Each 16KB W_out
// tile is async-staged into LDS once (double-buffered) and consumed by all
// 8 waves -> 8x less L2 traffic on the dominant (196 GFLOP) GEMM, i.e.
// 128 FLOP per L2 byte fetched.
// Grid: (SB/16/RG) * NSPLIT = 50 * 25 = 1250 blocks x 256 threads.
// ---------------------------------------------------------------------------
__global__ void loss_gemm_kernel(const __bf16* __restrict__ Hs,    // [SB,H]
                                 const __bf16* __restrict__ Wout,  // [V,H]
                                 const float*  __restrict__ bout,  // [V]
                                 const int*    __restrict__ tgt,   // [SB]
                                 float* __restrict__ pm,
                                 float* __restrict__ ps,
                                 float* __restrict__ pt) {
  __shared__ __align__(16) __bf16 Bs[2][16][HDIM];   // 2 x 16KB
  int tid  = threadIdx.x;
  int lane = tid & 31;
  int w    = tid >> 5;
  int rgi  = blockIdx.x / NSPLIT;
  int sp   = blockIdx.x % NSPLIT;
  int rb   = rgi * RG + w;                    // this wave's row-block

  // Cache this wave's A fragments (16 rows x K=512) in registers.
  v16bf areg[HDIM / 32];
#pragma unroll
  for (int kk = 0; kk < HDIM / 32; ++kk)
    areg[kk] = load_a16(Hs, HDIM, rb * 16, kk * 32, lane);

  int mh = (lane & 16) ? 8 : 0;
  int trow[8];
#pragma unroll
  for (int r = 0; r < 8; ++r) trow[r] = tgt[rb * 16 + mh + r];

  float rmax[8], rsum[8], rtg[8];
#pragma unroll
  for (int r = 0; r < 8; ++r) { rmax[r] = -3.0e38f; rsum[r] = 0.0f; rtg[r] = -3.0e38f; }

  unsigned lds0 = (unsigned)(unsigned long long)&Bs[0][0][0];
  unsigned lds1 = (unsigned)(unsigned long long)&Bs[1][0][0];

  // Prologue: stage tile 0 into buffer 0.
  async_stage_tile(Wout + (size_t)(sp * TPS) * 16 * HDIM, lds0, tid);
  wait_async0();
  __syncthreads();

  for (int tv = 0; tv < TPS; ++tv) {
    int cur = tv & 1;
    if (tv + 1 < TPS) {
      int v1 = (sp * TPS + tv + 1) * 16;
      async_stage_tile(Wout + (size_t)v1 * HDIM, cur ? lds0 : lds1, tid);
    }

    int v0 = (sp * TPS + tv) * 16;
    v8f acc = {};
#pragma unroll
    for (int kk = 0; kk < HDIM / 32; ++kk) {
      v16bf b = load_b16_lds(&Bs[cur][0][0], kk * 32, lane);
      acc = __builtin_amdgcn_wmma_f32_16x16x32_bf16(false, areg[kk], false, b,
                                                    (short)0, acc, false, false);
    }
    int v    = v0 + (lane & 15);
    float bo = bout[v];
#pragma unroll
    for (int r = 0; r < 8; ++r) {
      float val = acc[r] + bo;
      float nm  = fmaxf(rmax[r], val);
      rsum[r] = rsum[r] * __expf(rmax[r] - nm) + __expf(val - nm);
      rmax[r] = nm;
      if (v == trow[r]) rtg[r] = val;
    }

    wait_async0();       // next tile fully in LDS (this wave's share)
    __syncthreads();     // ... and everyone else's share too
  }

  // Reduce (max,sumexp,target) across the 16 lanes sharing each row.
  // xor masks 1,2,4,8 stay within each 16-lane half (bit4 untouched).
#pragma unroll
  for (int off = 8; off > 0; off >>= 1) {
#pragma unroll
    for (int r = 0; r < 8; ++r) {
      float om = __shfl_xor(rmax[r], off, 32);
      float os = __shfl_xor(rsum[r], off, 32);
      float ot = __shfl_xor(rtg[r],  off, 32);
      float nm = fmaxf(rmax[r], om);
      rsum[r] = rsum[r] * __expf(rmax[r] - nm) + os * __expf(om - nm);
      rmax[r] = nm;
      rtg[r]  = fmaxf(rtg[r], ot);
    }
  }

  if ((lane & 15) == 0) {
#pragma unroll
    for (int r = 0; r < 8; ++r) {
      int row = rb * 16 + mh + r;
      size_t idx = (size_t)row * NSPLIT + sp;
      pm[idx] = rmax[r];
      ps[idx] = rsum[r];
      pt[idx] = rtg[r];
    }
  }
}

// ---------------------------------------------------------------------------
// Combine split partials -> per-token masked NLL (ignore_index = PAD = 0)
// ---------------------------------------------------------------------------
__global__ void row_reduce_kernel(const float* __restrict__ pm,
                                  const float* __restrict__ ps,
                                  const float* __restrict__ pt,
                                  const int*   __restrict__ tgt,
                                  float* __restrict__ out_loss) {
  int i = blockIdx.x * blockDim.x + threadIdx.x;
  if (i >= SB) return;
  float M = -3.0e38f;
  for (int j = 0; j < NSPLIT; ++j) M = fmaxf(M, pm[(size_t)i * NSPLIT + j]);
  float Ssum = 0.0f, T = -3.0e38f;
  for (int j = 0; j < NSPLIT; ++j) {
    Ssum += ps[(size_t)i * NSPLIT + j] * __expf(pm[(size_t)i * NSPLIT + j] - M);
    T = fmaxf(T, pt[(size_t)i * NSPLIT + j]);
  }
  float lse = M + __logf(Ssum);
  out_loss[i] = (tgt[i] != 0) ? (lse - T) : 0.0f;
}

__global__ void obj_kernel(const float* __restrict__ out_loss,
                           const int* __restrict__ tgt,
                           float* __restrict__ obj) {
  __shared__ float ssum[256];
  __shared__ float scnt[256];
  float s = 0.0f, c = 0.0f;
  for (int i = threadIdx.x; i < SB; i += blockDim.x) {
    s += out_loss[i];
    c += (tgt[i] != 0) ? 1.0f : 0.0f;
  }
  ssum[threadIdx.x] = s;
  scnt[threadIdx.x] = c;
  __syncthreads();
  for (int off = 128; off > 0; off >>= 1) {
    if ((int)threadIdx.x < off) {
      ssum[threadIdx.x] += ssum[threadIdx.x + off];
      scnt[threadIdx.x] += scnt[threadIdx.x + off];
    }
    __syncthreads();
  }
  if (threadIdx.x == 0) obj[0] = ssum[0] / fmaxf(scnt[0], 1.0f);
}

// ---------------------------------------------------------------------------
// Host-side orchestration
// ---------------------------------------------------------------------------
extern "C" void kernel_launch(void* const* d_in, const int* in_sizes, int n_in,
                              void* d_out, int out_size, void* d_ws, size_t ws_size,
                              hipStream_t stream) {
  (void)in_sizes; (void)n_in; (void)out_size; (void)ws_size;
  // setup_inputs order:
  // 0 user, 1 item, 2 review_input, 3 review_target, 4 word_emb,
  // 5 W_ih, 6 W_hh, 7 b_ih, 8 b_hh, 9 W_out, 10 b_out
  const int*   rin   = (const int*)  d_in[2];
  const int*   rtgt  = (const int*)  d_in[3];
  const float* wemb  = (const float*)d_in[4];
  const float* Wih   = (const float*)d_in[5];
  const float* Whh   = (const float*)d_in[6];
  const float* bih   = (const float*)d_in[7];
  const float* bhh   = (const float*)d_in[8];
  const float* Wout  = (const float*)d_in[9];
  const float* bout  = (const float*)d_in[10];

  float* out_loss = (float*)d_out;         // [SB]
  float* obj      = out_loss + SB;         // [1]

  // Workspace carve-up (256B aligned).
  char* ws = (char*)d_ws;
  size_t off = 0;
  auto carve = [&](size_t bytes) -> char* {
    char* p = ws + off;
    off = (off + bytes + 255) & ~(size_t)255;
    return p;
  };
  __bf16* x_bf    = (__bf16*)carve((size_t)SB * DDIM * 2);
  __bf16* wih_bf  = (__bf16*)carve((size_t)H3 * DDIM * 2);
  __bf16* whh_bf  = (__bf16*)carve((size_t)H3 * HDIM * 2);
  __bf16* wout_bf = (__bf16*)carve((size_t)VOCAB * HDIM * 2);
  float*  GI      = (float*) carve((size_t)SB * H3 * 4);
  float*  h2[2];
  __bf16* hbf2[2];
  h2[0]   = (float*) carve((size_t)BATCH * HDIM * 4);
  h2[1]   = (float*) carve((size_t)BATCH * HDIM * 4);
  hbf2[0] = (__bf16*)carve((size_t)BATCH * HDIM * 2);
  hbf2[1] = (__bf16*)carve((size_t)BATCH * HDIM * 2);
  __bf16* hs_bf   = (__bf16*)carve((size_t)SB * HDIM * 2);
  float*  pm      = (float*) carve((size_t)SB * NSPLIT * 4);
  float*  ps      = (float*) carve((size_t)SB * NSPLIT * 4);
  float*  pt      = (float*) carve((size_t)SB * NSPLIT * 4);

  // --- Stage 0: precision conversion + embedding gather + h0 = 0 -----------
  {
    long n;
    n = (long)H3 * DDIM;
    cvt_bf16_kernel<<<(n + 255) / 256, 256, 0, stream>>>(Wih, wih_bf, n);
    n = (long)H3 * HDIM;
    cvt_bf16_kernel<<<(n + 255) / 256, 256, 0, stream>>>(Whh, whh_bf, n);
    n = (long)VOCAB * HDIM;
    cvt_bf16_kernel<<<(n + 255) / 256, 256, 0, stream>>>(Wout, wout_bf, n);
    n = (long)SB * DDIM;
    embed_kernel<<<(n + 255) / 256, 256, 0, stream>>>(rin, wemb, x_bf);
    zero_h_kernel<<<(BATCH * HDIM + 255) / 256, 256, 0, stream>>>(h2[0], hbf2[0]);
  }

  // --- Stage 1: GI = X @ W_ih^T + b_ih  (6400 x 1536, K=512) ---------------
  {
    int NT = H3 / 16;                       // 96
    int waves = (SB / 16) * NT;             // 38400
    gemm_bias_kernel<<<waves / 4, 128, 0, stream>>>(x_bf, wih_bf, bih, GI,
                                                    SB, H3, DDIM, NT);
  }

  // --- Stage 2: sequential GRU, one fused kernel per step ------------------
  for (int t = 0; t < S_LEN; ++t) {
    int rbuf = t & 1;
    int wbuf = rbuf ^ 1;
    gru_step_kernel<<<(BATCH / 16) * (HDIM / 16), 96, 0, stream>>>(
        GI, whh_bf, bhh, h2[rbuf], hbf2[rbuf], h2[wbuf], hbf2[wbuf], hs_bf, t);
  }

  // --- Stage 3: fused vocab GEMM + online softmax partials -----------------
  {
    int blocks = (SB / 16 / RG) * NSPLIT;   // 50 * 25 = 1250
    loss_gemm_kernel<<<blocks, LOSS_THREADS, 0, stream>>>(hs_bf, wout_bf, bout,
                                                          rtgt, pm, ps, pt);
  }

  // --- Stage 4: combine partials -> NLL per token, then masked mean --------
  row_reduce_kernel<<<(SB + 255) / 256, 256, 0, stream>>>(pm, ps, pt, rtgt,
                                                          out_loss);
  obj_kernel<<<1, 256, 0, stream>>>(out_loss, rtgt, obj);
}
